// SKMultiLoss_53626961658393
// MI455X (gfx1250) — matrix operand order repaired and checked by reading, not compiled, not run on hardware
//
#include <hip/hip_runtime.h>
#include <hip/hip_bf16.h>
#include <math.h>

typedef __attribute__((ext_vector_type(16))) __bf16 v16bf;
typedef __attribute__((ext_vector_type(8)))  __bf16 v8bf;
typedef __attribute__((ext_vector_type(8)))  float  v8f;

#define BATCH 16
#define NN    2048
#define KK    128
#define KP    132          // 128 data + 1 slack + 3 zero pad
#define EPS_INV 10.0       // 1/epsilon
#define LAM1 0.5f
#define RHO 0.5
#define FI_EXP 0.9090909090909090909   // gamma/(gamma+eps) = 1/1.1
#define SK_ITERS 1000
#define SK_TOL2 1e-12

// ---- CDNA5 async global->LDS copy (ASYNCcnt-tracked) ----
__device__ __forceinline__ void async_g2l_b32(void* lds, const void* g){
  unsigned l = (unsigned)(size_t)lds;           // low 32 bits = LDS offset
  asm volatile("global_load_async_to_lds_b32 %0, %1, off"
               :: "v"(l), "v"(g) : "memory");
}
__device__ __forceinline__ void wait_async0(){
#if __has_builtin(__builtin_amdgcn_s_wait_asynccnt)
  __builtin_amdgcn_s_wait_asynccnt(0);
#else
  asm volatile("s_wait_asynccnt 0" ::: "memory");
#endif
}

__device__ __forceinline__ unsigned pack_bf16(float lo, float hi){
  unsigned short a = __builtin_bit_cast(unsigned short, (__bf16)lo);
  unsigned short b = __builtin_bit_cast(unsigned short, (__bf16)hi);
  return (unsigned)a | ((unsigned)b << 16);
}

// C = -log_softmax(logits) over last dim (K=128). One wave per row.
__global__ __launch_bounds__(256) void log_softmax_kernel(
    const float* __restrict__ logits, float* __restrict__ C){
  int tid = threadIdx.x;
  int lane = tid & 31, wid = tid >> 5;
  int row = blockIdx.x * 8 + wid;
  const float* x = logits + (size_t)row * KK;
  float v[4]; float m = -1e30f;
  #pragma unroll
  for (int j=0;j<4;j++){ v[j] = x[lane + 32*j]; m = fmaxf(m, v[j]); }
  #pragma unroll
  for (int off=16; off>0; off>>=1) m = fmaxf(m, __shfl_xor(m, off, 32));
  float s = 0.f;
  #pragma unroll
  for (int j=0;j<4;j++) s += expf(v[j]-m);
  #pragma unroll
  for (int off=16; off>0; off>>=1) s += __shfl_xor(s, off, 32);
  float lse = m + logf(s);
  float* Cr = C + (size_t)row * KK;
  #pragma unroll
  for (int j=0;j<4;j++) Cr[lane + 32*j] = lse - v[j];
}

// Q0 = rho/(n*k), b0 = 1/(k+1)
__global__ void init_kernel(float* __restrict__ Q, double* __restrict__ b){
  size_t gid = (size_t)blockIdx.x * blockDim.x + threadIdx.x;
  if (gid < (size_t)BATCH*NN*KK) Q[gid] = (float)(RHO / ((double)NN * (double)KK));
  if (gid < (size_t)BATCH*129)   b[gid] = 1.0/129.0;
}

// cost = C - lam1*(S+S^T)@Q  ->  Qe = exp(-cost/eps), slack col = 1, pad = 0.
// 128x128 output tile per WG (8 waves x 16 rows). bf16 WMMA, f32 accumulate.
// S tiles staged via async global->LDS; bf16 tiles stored in fragment layout
// so each WMMA operand is two ds_load_b128.
__global__ __launch_bounds__(256) void cost_gemm_kernel(
    const float* __restrict__ S, const float* __restrict__ C,
    const float* __restrict__ Q, double* __restrict__ Qe){
  __shared__ float Sr[128][33];                                  // S row part (f32)
  __shared__ float St[32][132];                                  // S^T part (f32)
  __shared__ __attribute__((aligned(16))) __bf16 Asb[128][40];   // (S+S^T) bf16, row-major
  __shared__ __attribute__((aligned(16))) __bf16 Btb[128][40];   // Q bf16, transposed [n][k]
  int tid = threadIdx.x;
  int lane = tid & 31, wid = tid >> 5;
  int batch = blockIdx.x >> 4;
  int M0 = (blockIdx.x & 15) << 7;
  const float* Sb = S + (size_t)batch * NN * NN;
  const float* Qb = Q + (size_t)batch * NN * KK;

  v8f acc[8];
  #pragma unroll
  for (int t=0;t<8;t++)
    #pragma unroll
    for (int e=0;e<8;e++) acc[t][e] = 0.f;

  int h   = lane >> 4;    // lane half (K sub-block per WMMA striping)
  int l16 = lane & 15;

  for (int k0 = 0; k0 < NN; k0 += 32){
    // ---- async stage: S[M0+r][k0+c] and S[k0+c][M0+r] straight into LDS ----
    #pragma unroll
    for (int i=0;i<16;i++){
      int idx = tid + i*256;
      int r = idx >> 5, c = idx & 31;
      async_g2l_b32(&Sr[r][c], &Sb[(size_t)(M0 + r) * NN + (k0 + c)]);
    }
    #pragma unroll
    for (int i=0;i<16;i++){
      int idx = tid + i*256;
      int cc = idx >> 7, rr = idx & 127;
      async_g2l_b32(&St[cc][rr], &Sb[(size_t)(k0 + cc) * NN + (M0 + rr)]);
    }
    // ---- stage Q tile: f32 -> bf16, transposed to [n][k], packed b32 stores ----
    {
      int n = tid & 127;
      int kp0 = (tid >> 7) * 8;
      #pragma unroll
      for (int i=0;i<8;i++){
        int kp = kp0 + i, k = kp << 1;
        float lo = Qb[(size_t)(k0 + k)     * KK + n];
        float hi = Qb[(size_t)(k0 + k + 1) * KK + n];
        ((unsigned*)&Btb[n][0])[kp] = pack_bf16(lo, hi);
      }
    }
    wait_async0();
    __syncthreads();
    // ---- build bf16 A = S + S^T (pair-packed b32 stores) ----
    {
      int rbase = tid >> 4;          // 0..15
      int cp = tid & 15;             // column pair
      int c = cp << 1;
      #pragma unroll
      for (int i=0;i<8;i++){
        int r = rbase + i*16;
        float a0 = Sr[r][c]     + St[c][r];
        float a1 = Sr[r][c + 1] + St[c + 1][r];
        ((unsigned*)&Asb[r][0])[cp] = pack_bf16(a0, a1);
      }
    }
    __syncthreads();
    // ---- fragments: two ds_load_b128 per operand, then 8 WMMAs ----
    int m = (wid << 4) + l16;
    const v8bf* arow = (const v8bf*)&Asb[m][0];
    v8bf a0 = arow[h];        // k = 8h .. 8h+7
    v8bf a1 = arow[2 + h];    // k = 16+8h .. 16+8h+7
    v16bf af = __builtin_shufflevector(a0, a1,
        0,1,2,3,4,5,6,7,8,9,10,11,12,13,14,15);
    #pragma unroll
    for (int nt=0; nt<8; nt++){
      int n = (nt << 4) + l16;
      const v8bf* brow = (const v8bf*)&Btb[n][0];
      v8bf b0 = brow[h];        // k = 8h .. 8h+7
      v8bf b1 = brow[2 + h];    // k = 16+8h .. 16+8h+7
      v16bf bf = __builtin_shufflevector(b0, b1,
          0,1,2,3,4,5,6,7,8,9,10,11,12,13,14,15);
      acc[nt] = __builtin_amdgcn_wmma_f32_16x16x32_bf16(
          false, af, false, bf, (short)0, acc[nt], false, false);
    }
    __syncthreads();
  }

  // epilogue: cost -> Qe (fp64), plus slack/pad columns
  const float* Cb = C + (size_t)batch * NN * KK;
  double* Qeb = Qe + (size_t)batch * NN * KP;
  #pragma unroll
  for (int nt=0; nt<8; nt++){
    int n = (nt << 4) + l16;
    #pragma unroll
    for (int r=0;r<8;r++){
      int m = (wid << 4) + (h ? (r + 8) : r);   // C/D layout: lanes16-31 hold M=r+8
      int grow = M0 + m;
      float cost = Cb[(size_t)grow * KK + n] - LAM1 * acc[nt][r];
      Qeb[(size_t)grow * KP + n] = exp(-(double)cost * EPS_INV);
    }
  }
  if (lane < 16){
    int grow = M0 + (wid << 4) + lane;
    double* p = Qeb + (size_t)grow * KP;
    p[128] = 1.0; p[129] = 0.0; p[130] = 0.0; p[131] = 0.0;
  }
}

// Persistent per-batch Sinkhorn: fused a-step + column-sum (Qe read once/iter,
// L2-resident), deterministic LDS tree reductions, per-batch early exit.
__global__ __launch_bounds__(1024) void sinkhorn_kernel(
    const double* __restrict__ Qe, double* __restrict__ bglob,
    float* __restrict__ Q){
  __shared__ double b_s[KP];
  __shared__ double a_s[NN];
  __shared__ double wcol[32][KP];
  __shared__ double errparts[160];
  __shared__ double err_s;

  int tid = threadIdx.x;
  int lane = tid & 31, w = tid >> 5;
  int batch = blockIdx.x;
  const double* Qeb = Qe + (size_t)batch * NN * KP;
  double* bg = bglob + (size_t)batch * 129;

  if (tid < KP) b_s[tid] = (tid < 129) ? bg[tid] : 0.0;
  if (tid >= 132 && tid < 160) errparts[tid] = 0.0;
  __syncthreads();

  const double Pa = 1.0 / (double)NN;
  for (int it = 0; it < SK_ITERS; ++it){
    double part[5] = {0,0,0,0,0};
    for (int rr = 0; rr < 64; ++rr){
      int n = (w << 6) + rr;
      const double* qrow = Qeb + (size_t)n * KP;
      double q[5]; double dot = 0.0;
      #pragma unroll
      for (int j=0;j<5;j++){
        int c = lane + 32*j;
        double qq = (c < KP) ? qrow[c] : 0.0;
        double bb = (c < KP) ? b_s[c] : 0.0;
        q[j] = qq;
        dot += qq * bb;
      }
      #pragma unroll
      for (int off=16; off>0; off>>=1) dot += __shfl_xor(dot, off, 32);
      double a = Pa / dot;                       // a = Pa / (Qe @ b)
      if (lane == 0) a_s[n] = a;
      #pragma unroll
      for (int j=0;j<5;j++) part[j] += a * q[j]; // accumulate Qe^T @ a
    }
    #pragma unroll
    for (int j=0;j<5;j++){
      int c = lane + 32*j;
      if (c < KP) wcol[w][c] = part[j];
    }
    __syncthreads();
    if (tid < KP){
      double s = 0.0;
      #pragma unroll 8
      for (int ww=0; ww<32; ++ww) s += wcol[ww][tid];   // deterministic reduce
      double bb;
      if (tid < 129){
        double Pb = (tid < 128) ? (RHO / (double)KK) : (1.0 - RHO);
        bb = Pb / s;
        if (tid < 128) bb = pow(bb, FI_EXP);            // semi-relaxed KL damping
      } else bb = 0.0;
      double d = bb - b_s[tid];
      errparts[tid] = (tid < 129) ? d * d : 0.0;
      b_s[tid] = bb;
    }
    __syncthreads();
    if (w == 0){
      double e = 0.0;
      #pragma unroll
      for (int j=0;j<5;j++){ int c = lane + 32*j; if (c < 160) e += errparts[c]; }
      #pragma unroll
      for (int off=16; off>0; off>>=1) e += __shfl_xor(e, off, 32);
      if (lane == 0) err_s = e;
    }
    __syncthreads();
    if (err_s <= SK_TOL2) break;   // uniform: read after barrier
  }

  // plan = a * Qe * b^T, drop slack column
  float* Qb = Q + (size_t)batch * NN * KK;
  for (int rr = 0; rr < 64; ++rr){
    int n = (w << 6) + rr;
    const double* qrow = Qeb + (size_t)n * KP;
    double a = a_s[n];
    #pragma unroll
    for (int j=0;j<4;j++){
      int c = lane + 32*j;
      Qb[(size_t)n * KK + c] = (float)(a * qrow[c] * b_s[c]);
    }
  }
  if (tid < 129) bg[tid] = b_s[tid];
}

__global__ void finalize_kernel(const float* __restrict__ Q, float* __restrict__ out){
  size_t gid = (size_t)blockIdx.x * blockDim.x + threadIdx.x;
  if (gid < (size_t)BATCH*NN*KK) out[gid] = Q[gid] * (float)NN;
}

extern "C" void kernel_launch(void* const* d_in, const int* in_sizes, int n_in,
                              void* d_out, int out_size, void* d_ws, size_t ws_size,
                              hipStream_t stream){
  (void)in_sizes; (void)n_in; (void)out_size; (void)ws_size;
  const float* logits = (const float*)d_in[0];
  const float* S      = (const float*)d_in[1];

  // ws layout: C[f32 16MB] | Q[f32 16MB] | Qe[f64 34.6MB] | b[f64 16.5KB]
  float*  Cw = (float*)d_ws;
  float*  Qw = Cw + (size_t)BATCH*NN*KK;
  double* Qe = (double*)((char*)d_ws + 2ull * BATCH * NN * KK * 4ull);
  double* bw = (double*)((char*)d_ws + 2ull * BATCH * NN * KK * 4ull
                                     + (size_t)BATCH * NN * KP * 8ull);

  log_softmax_kernel<<<(BATCH*NN)/8, 256, 0, stream>>>(logits, Cw);
  init_kernel<<<(BATCH*NN*KK)/256, 256, 0, stream>>>(Qw, bw);
  for (int outer = 0; outer < 10; ++outer){
    cost_gemm_kernel<<<BATCH*(NN/128), 256, 0, stream>>>(S, Cw, Qw, Qe);
    sinkhorn_kernel<<<BATCH, 1024, 0, stream>>>(Qe, bw, Qw);
  }
  finalize_kernel<<<(BATCH*NN*KK)/256, 256, 0, stream>>>(Qw, (float*)d_out);
}
